// LinkPredictionSRHGNPlus_33294586479050
// MI455X (gfx1250) — compile-verified, matrix operand order checked by reading, not run on hardware
//
#include <hip/hip_runtime.h>
#include <hip/hip_bf16.h>
#include <math.h>

#define DD 128
#define HNn 4
#define HTn 4
#define WSTRIDE 136   // 128 + 8 bf16 pad: rotates LDS banks by 4 per column, keeps 16B align

typedef __attribute__((ext_vector_type(16))) __bf16 v16bf;
typedef __attribute__((ext_vector_type(8)))  __bf16 v8bf;
typedef __attribute__((ext_vector_type(8)))  float  v8f;

__device__ __forceinline__ float gelu_exact(float x) {
  return 0.5f * x * (1.0f + erff(x * 0.70710678118654752f));
}
__device__ __forceinline__ float lrelu(float x) { return x > 0.0f ? x : 0.01f * x; }

__device__ __forceinline__ float wave_sum(float v) {
#pragma unroll
  for (int o = 16; o > 0; o >>= 1) v += __shfl_xor(v, o, 32);
  return v;
}

__device__ __forceinline__ void atomicMaxFloat(float* addr, float val) {
  // IEEE-754 ordered-int trick (works for +/- including -inf init)
  if (val >= 0.0f) atomicMax((int*)addr, __float_as_int(val));
  else             atomicMin((unsigned int*)addr, (unsigned int)__float_as_int(val));
}

// ---------------------------------------------------------------------------
// Y[N,128] = act(X[N,128] @ W[128,128] + bias)    (WMMA bf16 -> f32 acc)
// One block = 256 threads = 8 waves; each wave computes 16 rows x 128 cols.
// W is transposed into LDS as bf16 (Wt[n][k]) so every B-fragment half is one
// 16B-aligned ds_load_b128. A-fragments are 4 global_load_b128 per K-step,
// double-buffered: the next step's loads issue before the current step's
// WMMAs so global latency hides behind matrix ops. OOB rows clamp to a valid
// row; their products land only in D rows that are never stored.
// ---------------------------------------------------------------------------
__global__ __launch_bounds__(256)
void gemm128_bias_act(const float* __restrict__ X, const float* __restrict__ W,
                      const float* __restrict__ bias, float* __restrict__ Y,
                      int N, int act)
{
  __shared__ __bf16 Wt[DD * WSTRIDE];
  const int t = threadIdx.x;
  for (int i = t; i < DD * DD; i += 256) {
    const int k = i >> 7, n = i & 127;
    Wt[n * WSTRIDE + k] = (__bf16)W[i];
  }
  __syncthreads();

  const int wave = t >> 5;
  const int lane = t & 31;
  const int half = lane >> 4;     // K half selector for A/B fragments
  const int m    = lane & 15;     // row (A) / col (B) within 16-tile
  const int rowBase = blockIdx.x * 128 + wave * 16;
  int row = rowBase + m;
  if (row >= N) row = N - 1;      // clamp: OOB rows computed but never stored
  const float* __restrict__ xr = X + (size_t)row * DD;

  v8f acc[8] = {};

  auto loadA = [&](int k0, float4* dst) {
    const int kA = k0 + half * 8;        // elements 0..7  -> K = kA+e
    const int kB = k0 + 16 + half * 8;   // elements 8..15 -> K = kB+e
    dst[0] = *(const float4*)(xr + kA);
    dst[1] = *(const float4*)(xr + kA + 4);
    dst[2] = *(const float4*)(xr + kB);
    dst[3] = *(const float4*)(xr + kB + 4);
  };

  float4 abuf[2][4];
  loadA(0, abuf[0]);

#pragma unroll
  for (int s = 0; s < 4; ++s) {
    const int k0 = s * 32;
    if (s < 3) loadA(k0 + 32, abuf[(s + 1) & 1]);   // prefetch next K-step
    const float4* af4 = abuf[s & 1];
    v16bf a;
#pragma unroll
    for (int q = 0; q < 4; ++q) {
      a[q * 4 + 0] = (__bf16)af4[q].x;
      a[q * 4 + 1] = (__bf16)af4[q].y;
      a[q * 4 + 2] = (__bf16)af4[q].z;
      a[q * 4 + 3] = (__bf16)af4[q].w;
    }
    const int kA = k0 + half * 8;
    const int kB = k0 + 16 + half * 8;
#pragma unroll
    for (int nt = 0; nt < 8; ++nt) {
      const int n = nt * 16 + m;
      const v8bf blo = *(const v8bf*)(Wt + n * WSTRIDE + kA);
      const v8bf bhi = *(const v8bf*)(Wt + n * WSTRIDE + kB);
      const v16bf b = __builtin_shufflevector(blo, bhi,
          0, 1, 2, 3, 4, 5, 6, 7, 8, 9, 10, 11, 12, 13, 14, 15);
      acc[nt] = __builtin_amdgcn_wmma_f32_16x16x32_bf16(
          false, a, false, b, (short)0, acc[nt], false, false);
    }
  }

  // C/D layout: VGPR j of lane -> M = j + 8*half, N = lane&15
#pragma unroll
  for (int nt = 0; nt < 8; ++nt) {
    const int col = nt * 16 + m;
    const float bv = bias[col];
#pragma unroll
    for (int j = 0; j < 8; ++j) {
      const int r = rowBase + half * 8 + j;
      if (r < N) {
        float v = acc[nt][j] + bv;
        if (act == 1) v = gelu_exact(v);
        Y[(size_t)r * DD + col] = v;
      }
    }
  }
}

// ---------------------------------------------------------------------------
// Y[row*ystride + yoff + h] (+)= (l2n?)X[row,:] @ W[128,4] + b[h]
// One wave per row, 4 activations per lane.
// ---------------------------------------------------------------------------
__global__ __launch_bounds__(256)
void proj_small(const float* __restrict__ X, const float* __restrict__ W,
                const float* __restrict__ b, float* __restrict__ Y,
                int N, int ystride, int yoff, int accum, int l2in)
{
  const int wid  = (int)((blockIdx.x * blockDim.x + threadIdx.x) >> 5);
  const int lane = threadIdx.x & 31;
  if (wid >= N) return;
  const float* xr = X + (size_t)wid * DD;
  float x[4];
#pragma unroll
  for (int j = 0; j < 4; ++j) x[j] = xr[lane + 32 * j];
  if (l2in) {
    float ss = wave_sum(x[0]*x[0] + x[1]*x[1] + x[2]*x[2] + x[3]*x[3]);
    float inv = 1.0f / fmaxf(sqrtf(ss), 1e-9f);
#pragma unroll
    for (int j = 0; j < 4; ++j) x[j] *= inv;
  }
#pragma unroll
  for (int h = 0; h < 4; ++h) {
    float s = 0.0f;
#pragma unroll
    for (int j = 0; j < 4; ++j) s += x[j] * W[(lane + 32 * j) * 4 + h];
    s = wave_sum(s);
    if (lane == 0) {
      float v = s + b[h];
      float* dst = Y + (size_t)wid * ystride + yoff + h;
      if (accum) *dst += v; else *dst = v;
    }
  }
}

__global__ __launch_bounds__(256)
void l2norm_rows(float* __restrict__ X, int N)
{
  const int wid  = (int)((blockIdx.x * blockDim.x + threadIdx.x) >> 5);
  const int lane = threadIdx.x & 31;
  if (wid >= N) return;
  float* xr = X + (size_t)wid * DD;
  float x[4];
#pragma unroll
  for (int j = 0; j < 4; ++j) x[j] = xr[lane + 32 * j];
  float ss = wave_sum(x[0]*x[0] + x[1]*x[1] + x[2]*x[2] + x[3]*x[3]);
  float inv = 1.0f / fmaxf(sqrtf(ss), 1e-9f);
#pragma unroll
  for (int j = 0; j < 4; ++j) xr[lane + 32 * j] = x[j] * inv;
}

__global__ __launch_bounds__(256)
void fill_f32(float* __restrict__ p, float v, long long n)
{
  long long i = (long long)blockIdx.x * blockDim.x + threadIdx.x;
  if (i < n) p[i] = v;
}

// ---------------------------------------------------------------------------
// Edge softmax (3 passes) + weighted scatter
// ---------------------------------------------------------------------------
__global__ __launch_bounds__(256)
void edge_logits_max(const float* __restrict__ sa, const float* __restrict__ da,
                     const int* __restrict__ u, const int* __restrict__ v,
                     float* __restrict__ logit, float* __restrict__ mx, int E)
{
  int i = blockIdx.x * blockDim.x + threadIdx.x;
  int e = i >> 2, h = i & 3;
  if (e >= E) return;
  float x = lrelu(sa[(size_t)u[e] * HNn + h] + da[(size_t)v[e] * HNn + h]);
  logit[(size_t)e * HNn + h] = x;
  atomicMaxFloat(&mx[(size_t)v[e] * HNn + h], x);
}

__global__ __launch_bounds__(256)
void edge_exp_sum(float* __restrict__ logit, const int* __restrict__ v,
                  const float* __restrict__ mx, float* __restrict__ sm, int E)
{
  int i = blockIdx.x * blockDim.x + threadIdx.x;
  int e = i >> 2, h = i & 3;
  if (e >= E) return;
  float ex = expf(logit[(size_t)e * HNn + h] - mx[(size_t)v[e] * HNn + h]);
  logit[(size_t)e * HNn + h] = ex;
  atomicAdd(&sm[(size_t)v[e] * HNn + h], ex);
}

__global__ __launch_bounds__(256)
void edge_scatter(const float* __restrict__ hs, const float* __restrict__ logit,
                  const float* __restrict__ sm, const int* __restrict__ u,
                  const int* __restrict__ v, float* __restrict__ z, int E)
{
  long long i = (long long)blockIdx.x * blockDim.x + threadIdx.x;
  int e = (int)(i >> 7), c = (int)(i & 127);
  if (e >= E) return;
  int h = c >> 5;  // D/HN = 32
  float w = logit[(size_t)e * HNn + h] / sm[(size_t)v[e] * HNn + h];
  atomicAdd(&z[(size_t)v[e] * DD + c], hs[(size_t)u[e] * DD + c] * w);
}

// rel[a,h] = rel_emb[l,a,:] @ rW[l,a,:,h] + rb[l,a,h]  for a=0..3
__global__ void rel_compute(const float* __restrict__ rel_emb_l,
                            const float* __restrict__ rW_l,
                            const float* __restrict__ rb_l,
                            float* __restrict__ out)
{
  int t = threadIdx.x;
  if (t >= 16) return;
  int a = t >> 2, h = t & 3;
  float s = 0.0f;
  for (int k = 0; k < DD; ++k)
    s += rel_emb_l[a * DD + k] * rW_l[((size_t)a * DD + k) * HTn + h];
  out[a * HTn + h] = s + rb_l[a * HTn + h];
}

// hnew = l2n(gelu(sum_r z_r * attn_r + h)); attn = 0.5*softmax_r(lrelu(sem)) + 0.5*softmax_r(lrelu(rel))
__global__ __launch_bounds__(256)
void agg_combine(const float* __restrict__ z0, const float* __restrict__ z1,
                 const float* __restrict__ sem, const float* __restrict__ relv,
                 const float* __restrict__ h, float* __restrict__ hnew, int N)
{
  const int wid  = (int)((blockIdx.x * blockDim.x + threadIdx.x) >> 5);
  const int lane = threadIdx.x & 31;
  if (wid >= N) return;
  float attn0[4], attn1[4];
#pragma unroll
  for (int hh = 0; hh < 4; ++hh) {
    float r0 = lrelu(relv[hh]),      r1 = lrelu(relv[4 + hh]);
    float rm = fmaxf(r0, r1);
    float e0 = expf(r0 - rm), e1 = expf(r1 - rm);
    float rs = e0 + e1;
    float s0 = lrelu(sem[(size_t)wid * 8 + hh]);
    float s1 = lrelu(sem[(size_t)wid * 8 + 4 + hh]);
    float sm2 = fmaxf(s0, s1);
    float f0 = expf(s0 - sm2), f1 = expf(s1 - sm2);
    float fs = f0 + f1;
    attn0[hh] = 0.5f * (f0 / fs) + 0.5f * (e0 / rs);
    attn1[hh] = 0.5f * (f1 / fs) + 0.5f * (e1 / rs);
  }
  float tbuf[4], ss = 0.0f;
#pragma unroll
  for (int j = 0; j < 4; ++j) {      // col = lane + 32*j -> head j (D/HT = 32)
    int c = lane + 32 * j;
    float pre = z0[(size_t)wid * DD + c] * attn0[j]
              + z1[(size_t)wid * DD + c] * attn1[j]
              + h[(size_t)wid * DD + c];
    float g = gelu_exact(pre);
    tbuf[j] = g;
    ss += g * g;
  }
  ss = wave_sum(ss);
  float inv = 1.0f / fmaxf(sqrtf(ss), 1e-9f);
#pragma unroll
  for (int j = 0; j < 4; ++j)
    hnew[(size_t)wid * DD + lane + 32 * j] = tbuf[j] * inv;
}

__global__ __launch_bounds__(256)
void pair_dot(const float* __restrict__ x, const int* __restrict__ pi,
              const int* __restrict__ pj, float* __restrict__ out, int P)
{
  const int wid  = (int)((blockIdx.x * blockDim.x + threadIdx.x) >> 5);
  const int lane = threadIdx.x & 31;
  if (wid >= P) return;
  const float* xi = x + (size_t)pi[wid] * DD;
  const float* xj = x + (size_t)pj[wid] * DD;
  float s = 0.0f;
#pragma unroll
  for (int j = 0; j < 4; ++j) s += xi[lane + 32 * j] * xj[lane + 32 * j];
  s = wave_sum(s);
  if (lane == 0) out[wid] = s;
}

// ---------------------------------------------------------------------------
extern "C" void kernel_launch(void* const* d_in, const int* in_sizes, int n_in,
                              void* d_out, int out_size, void* d_ws, size_t ws_size,
                              hipStream_t stream)
{
  (void)in_sizes; (void)n_in; (void)out_size; (void)ws_size;
  const int NA = 20000, NP = 40000, E = 200000, P = 100000, L = 2;

  const float* feat_a  = (const float*)d_in[0];
  const float* feat_p  = (const float*)d_in[1];
  const float* preW_a  = (const float*)d_in[2];
  const float* preb_a  = (const float*)d_in[3];
  const float* preW_p  = (const float*)d_in[4];
  const float* preb_p  = (const float*)d_in[5];
  const float* nodeW   = (const float*)d_in[6];
  const float* nodeb   = (const float*)d_in[7];
  const float* edgeW   = (const float*)d_in[8];
  const float* edgeb   = (const float*)d_in[9];
  const float* saW     = (const float*)d_in[10];
  const float* sab     = (const float*)d_in[11];
  const float* daW     = (const float*)d_in[12];
  const float* dab     = (const float*)d_in[13];
  const float* ssW     = (const float*)d_in[14];
  const float* ssb     = (const float*)d_in[15];
  const float* sdW     = (const float*)d_in[16];
  const float* sdb     = (const float*)d_in[17];
  const float* rW      = (const float*)d_in[18];
  const float* rb      = (const float*)d_in[19];
  const float* rel_emb = (const float*)d_in[20];
  const int* writes_src = (const int*)d_in[21];
  const int* writes_dst = (const int*)d_in[22];
  const int* cites_src  = (const int*)d_in[23];
  const int* cites_dst  = (const int*)d_in[24];
  const int* wb_src     = (const int*)d_in[25];
  const int* wb_dst     = (const int*)d_in[26];
  const int* co_src     = (const int*)d_in[27];
  const int* co_dst     = (const int*)d_in[28];
  const int* pair_i     = (const int*)d_in[29];
  const int* pair_j     = (const int*)d_in[30];

  float* ws = (float*)d_ws;
  size_t off = 0;
  auto alloc = [&](size_t n) { float* p = ws + off; off += (n + 63) & ~(size_t)63; return p; };

  float* hA[2] = { alloc((size_t)NA * DD), alloc((size_t)NA * DD) };
  float* hP[2] = { alloc((size_t)NP * DD), alloc((size_t)NP * DD) };
  float* hd_a  = alloc((size_t)NA * DD);   // becomes l2n'd (nd_a) after msg passes
  float* hd_p  = alloc((size_t)NP * DD);
  float* z_a0  = alloc((size_t)NA * DD);
  float* z_a1  = alloc((size_t)NA * DD);
  float* z_p0  = alloc((size_t)NP * DD);
  float* z_p1  = alloc((size_t)NP * DD);
  float* hs    = alloc((size_t)NP * DD);   // per-relation source transform (max size)
  float* sa    = alloc((size_t)NP * HNn);
  float* da_a0 = alloc((size_t)NA * HNn);
  float* da_a1 = alloc((size_t)NA * HNn);
  float* da_p0 = alloc((size_t)NP * HNn);
  float* da_p1 = alloc((size_t)NP * HNn);
  float* logit = alloc((size_t)E * HNn);
  float* mx    = alloc((size_t)NP * HNn);
  float* sm    = alloc((size_t)NP * HNn);
  float* sem_a = alloc((size_t)NA * 2 * HTn);
  float* sem_p = alloc((size_t)NP * 2 * HTn);
  float* relv  = alloc(64);

  auto gemm = [&](const float* X, const float* W, const float* b, float* Y, int N, int act) {
    gemm128_bias_act<<<dim3((unsigned)((N + 127) / 128)), dim3(256), 0, stream>>>(X, W, b, Y, N, act);
  };
  auto proj = [&](const float* X, const float* W, const float* b, float* Y, int N,
                  int ystride, int yoff, int accum, int l2in) {
    proj_small<<<dim3((unsigned)((N + 7) / 8)), dim3(256), 0, stream>>>(X, W, b, Y, N, ystride, yoff, accum, l2in);
  };
  auto fill = [&](float* p, float v, size_t n) {
    fill_f32<<<dim3((unsigned)((n + 255) / 256)), dim3(256), 0, stream>>>(p, v, (long long)n);
  };

  // pre-transform: h = gelu(feat @ preW + preb)
  gemm(feat_a, preW_a, preb_a, hA[0], NA, 1);
  gemm(feat_p, preW_p, preb_p, hP[0], NP, 1);

  int cur = 0;
  for (int l = 0; l < L; ++l) {
    // hd = h @ nodeW[l,d] + nodeb[l,d]   (shared between msg() and agg())
    gemm(hA[cur], nodeW + (size_t)(l * 2 + 0) * DD * DD, nodeb + (size_t)(l * 2 + 0) * DD, hd_a, NA, 0);
    gemm(hP[cur], nodeW + (size_t)(l * 2 + 1) * DD * DD, nodeb + (size_t)(l * 2 + 1) * DD, hd_p, NP, 0);

    // destination attention scores per relation (e: 0 writes, 1 cites, 2 wb, 3 co)
    proj(hd_a, daW + (size_t)(l * 4 + 2) * DD * HNn, dab + (size_t)(l * 4 + 2) * HNn, da_a0, NA, HNn, 0, 0, 0);
    proj(hd_a, daW + (size_t)(l * 4 + 3) * DD * HNn, dab + (size_t)(l * 4 + 3) * HNn, da_a1, NA, HNn, 0, 0, 0);
    proj(hd_p, daW + (size_t)(l * 4 + 0) * DD * HNn, dab + (size_t)(l * 4 + 0) * HNn, da_p0, NP, HNn, 0, 0, 0);
    proj(hd_p, daW + (size_t)(l * 4 + 1) * DD * HNn, dab + (size_t)(l * 4 + 1) * HNn, da_p1, NP, HNn, 0, 0, 0);

    fill(z_a0, 0.0f, (size_t)NA * DD);
    fill(z_a1, 0.0f, (size_t)NA * DD);
    fill(z_p0, 0.0f, (size_t)NP * DD);
    fill(z_p1, 0.0f, (size_t)NP * DD);

    auto run_rel = [&](const float* hsrc, int Ns, int e, const int* u, const int* v,
                       const float* dsc, int n, float* z) {
      gemm(hsrc, edgeW + (size_t)(l * 4 + e) * DD * DD, edgeb + (size_t)(l * 4 + e) * DD, hs, Ns, 0);
      proj(hs, saW + (size_t)(l * 4 + e) * DD * HNn, sab + (size_t)(l * 4 + e) * HNn, sa, Ns, HNn, 0, 0, 0);
      fill(mx, -INFINITY, (size_t)n * HNn);
      fill(sm, 0.0f, (size_t)n * HNn);
      edge_logits_max<<<dim3((unsigned)((E * HNn + 255) / 256)), dim3(256), 0, stream>>>(sa, dsc, u, v, logit, mx, E);
      edge_exp_sum  <<<dim3((unsigned)((E * HNn + 255) / 256)), dim3(256), 0, stream>>>(logit, v, mx, sm, E);
      edge_scatter  <<<dim3((unsigned)(((size_t)E * DD + 255) / 256)), dim3(256), 0, stream>>>(hs, logit, sm, u, v, z, E);
    };
    run_rel(hP[cur], NP, 2, wb_src,     wb_dst,     da_a0, NA, z_a0);  // written_by p->a
    run_rel(hA[cur], NA, 3, co_src,     co_dst,     da_a1, NA, z_a1);  // coauthor   a->a
    run_rel(hA[cur], NA, 0, writes_src, writes_dst, da_p0, NP, z_p0);  // writes     a->p
    run_rel(hP[cur], NP, 1, cites_src,  cites_dst,  da_p1, NP, z_p1);  // cites      p->p

    // nd = l2n(hd) in place (da scores already extracted)
    l2norm_rows<<<dim3((unsigned)((NA + 7) / 8)), dim3(256), 0, stream>>>(hd_a, NA);
    l2norm_rows<<<dim3((unsigned)((NP + 7) / 8)), dim3(256), 0, stream>>>(hd_p, NP);

    // semantic logits sem[:, r, :] = l2n(z_r)@ssW + ssb + nd@sdW + sdb (attn ids a: author 0,1; paper 2,3)
    proj(z_a0, ssW + (size_t)(l * 4 + 0) * DD * HTn, ssb + (size_t)(l * 4 + 0) * HTn, sem_a, NA, 2 * HTn, 0,   0, 1);
    proj(hd_a, sdW + (size_t)(l * 4 + 0) * DD * HTn, sdb + (size_t)(l * 4 + 0) * HTn, sem_a, NA, 2 * HTn, 0,   1, 0);
    proj(z_a1, ssW + (size_t)(l * 4 + 1) * DD * HTn, ssb + (size_t)(l * 4 + 1) * HTn, sem_a, NA, 2 * HTn, HTn, 0, 1);
    proj(hd_a, sdW + (size_t)(l * 4 + 1) * DD * HTn, sdb + (size_t)(l * 4 + 1) * HTn, sem_a, NA, 2 * HTn, HTn, 1, 0);
    proj(z_p0, ssW + (size_t)(l * 4 + 2) * DD * HTn, ssb + (size_t)(l * 4 + 2) * HTn, sem_p, NP, 2 * HTn, 0,   0, 1);
    proj(hd_p, sdW + (size_t)(l * 4 + 2) * DD * HTn, sdb + (size_t)(l * 4 + 2) * HTn, sem_p, NP, 2 * HTn, 0,   1, 0);
    proj(z_p1, ssW + (size_t)(l * 4 + 3) * DD * HTn, ssb + (size_t)(l * 4 + 3) * HTn, sem_p, NP, 2 * HTn, HTn, 0, 1);
    proj(hd_p, sdW + (size_t)(l * 4 + 3) * DD * HTn, sdb + (size_t)(l * 4 + 3) * HTn, sem_p, NP, 2 * HTn, HTn, 1, 0);

    rel_compute<<<dim3(1), dim3(32), 0, stream>>>(rel_emb + (size_t)l * 4 * DD,
                                                  rW + (size_t)l * 4 * DD * HTn,
                                                  rb + (size_t)l * 4 * HTn, relv);

    const int nxt = cur ^ 1;
    agg_combine<<<dim3((unsigned)((NA + 7) / 8)), dim3(256), 0, stream>>>(z_a0, z_a1, sem_a, relv + 0, hA[cur], hA[nxt], NA);
    agg_combine<<<dim3((unsigned)((NP + 7) / 8)), dim3(256), 0, stream>>>(z_p0, z_p1, sem_p, relv + 8, hP[cur], hP[nxt], NP);
    cur = nxt;
  }

  pair_dot<<<dim3((unsigned)((P + 7) / 8)), dim3(256), 0, stream>>>(hA[cur], pair_i, pair_j, (float*)d_out, P);
}
